// MyMultiLayerGRU_72121090835055
// MI455X (gfx1250) — compile-verified, matrix operand order checked by reading, not run on hardware
//
#include <hip/hip_runtime.h>
#include <hip/hip_bf16.h>

typedef __attribute__((ext_vector_type(16))) __bf16 v16bf;
typedef __attribute__((ext_vector_type(8)))  float  v8f;

#define B_SZ 64
#define T_SZ 512
#define HID  512
#define IN0  256

union Frag {
  v16bf v;
  uint4 q[2];
};

__device__ __forceinline__ unsigned short f2bf(float f) {
  unsigned int u = __float_as_uint(f);
  unsigned int r = u + 0x7FFFu + ((u >> 16) & 1u);   // round-to-nearest-even
  return (unsigned short)(r >> 16);
}

__device__ __forceinline__ float sigmoidf_fast(float x) {
  return 1.0f / (1.0f + __expf(-x));
}
__device__ __forceinline__ float tanhf_fast(float x) {
  return 1.0f - 2.0f / (__expf(2.0f * x) + 1.0f);
}

// ---------------------------------------------------------------------------
// Pack fp32 weights into bf16, WMMA-B fragment-ready layout.
// For a KxN matrix tiled into (32x16) WMMA-B tiles, tile (kb,nb) is stored as
// 512 contiguous bf16: element index = lane*16 + j, where lane owns column
// n = nb*16 + (lane&15) and j maps to k-local:
//   j<8  -> k = (lane>>4)*8 + j
//   j>=8 -> k = 16 + (lane>>4)*8 + (j-8)
// Bzr: K=(in+HID), N=1024 ([z-gate | r-gate] cols), rows stacked [W ; U].
// Bh : K=(in+HID), N=512, rows stacked [Wh ; Uh].
// ---------------------------------------------------------------------------
__global__ void pack_weights_kernel(const float* __restrict__ Wz, const float* __restrict__ Uz,
                                    const float* __restrict__ Wr, const float* __restrict__ Ur,
                                    const float* __restrict__ Wh, const float* __restrict__ Uh,
                                    int in_dim,
                                    unsigned short* __restrict__ Bzr,
                                    unsigned short* __restrict__ Bh) {
  const int K = in_dim + HID;
  const long long totZR = (long long)K * 1024;
  const long long totH  = (long long)K * 512;
  long long idx = (long long)blockIdx.x * blockDim.x + threadIdx.x;
  if (idx >= totZR + totH) return;
  const bool isH = (idx >= totZR);
  long long id = isH ? (idx - totZR) : idx;
  const int e = (int)(id & 511);
  const long long frag = id >> 9;
  const int lane = e >> 4, j = e & 15;
  const int hlf = lane >> 4, col = lane & 15;
  const int kl = (j < 8) ? (hlf * 8 + j) : (16 + hlf * 8 + (j - 8));
  int kb, nb;
  if (isH) { kb = (int)(frag >> 5); nb = (int)(frag & 31); }
  else     { kb = (int)(frag >> 6); nb = (int)(frag & 63); }
  const int k = kb * 32 + kl;
  const int n = nb * 16 + col;
  float v;
  if (isH) {
    v = (k < in_dim) ? Wh[(size_t)k * HID + n] : Uh[(size_t)(k - in_dim) * HID + n];
    Bh[id] = f2bf(v);
  } else {
    if (n < HID) {
      v = (k < in_dim) ? Wz[(size_t)k * HID + n] : Uz[(size_t)(k - in_dim) * HID + n];
    } else {
      const int n2 = n - HID;
      v = (k < in_dim) ? Wr[(size_t)k * HID + n2] : Ur[(size_t)(k - in_dim) * HID + n2];
    }
    Bzr[id] = f2bf(v);
  }
}

__global__ void convert_x_kernel(const float* __restrict__ src,
                                 unsigned short* __restrict__ dst, int n) {
  int i = blockIdx.x * blockDim.x + threadIdx.x;
  if (i < n) dst[i] = f2bf(src[i]);
}

__global__ void zero_h_kernel(float* __restrict__ hf, unsigned short* __restrict__ hb) {
  int i = blockIdx.x * blockDim.x + threadIdx.x;
  if (i < B_SZ * HID) { hf[i] = 0.0f; hb[i] = 0; }
}

// ---------------------------------------------------------------------------
// Phase 1 per timestep: [x_t , h] (64 x (in+512)) @ Bzr ((in+512) x 1024)
//   cols [0,512)  -> z = sigmoid(. + bz)           (stored fp32)
//   cols [512,1K) -> r = sigmoid(. + br); rh = r*h (stored bf16)
// 4 M-tiles x 64 N-tiles; 64 blocks x 4 waves; one 16x16 WMMA tile per wave.
// ---------------------------------------------------------------------------
__global__ __launch_bounds__(128)
void gru_phase1_kernel(const unsigned short* __restrict__ xin,
                       const unsigned short* __restrict__ h_bf,
                       const float* __restrict__ h_f32,
                       const unsigned short* __restrict__ Bzr,
                       const float* __restrict__ bz,
                       const float* __restrict__ br,
                       float* __restrict__ zbuf,
                       unsigned short* __restrict__ rhbuf,
                       int t, int in_dim) {
  const int lane = threadIdx.x & 31;
  const int w = blockIdx.x * 4 + (threadIdx.x >> 5);
  const int mt = w & 3;
  const int nt = w >> 2;
  const int hlf = lane >> 4;
  const int l15 = lane & 15;
  const int arow = mt * 16 + l15;
  const unsigned short* xrow = xin + ((size_t)arow * T_SZ + t) * in_dim;
  const unsigned short* hrow = h_bf + (size_t)arow * HID;

  v8f acc = {};
  const int kbx = in_dim >> 5;
  const int kbt = (in_dim + HID) >> 5;
#pragma unroll 4
  for (int kb = 0; kb < kbx; ++kb) {          // x_t part of A
    Frag a, b;
    const uint4* ap = (const uint4*)(xrow + kb * 32 + hlf * 8);
    a.q[0] = ap[0]; a.q[1] = ap[2];
    const uint4* bp = (const uint4*)(Bzr + (((size_t)kb * 64 + nt) << 9) + (lane << 4));
    b.q[0] = bp[0]; b.q[1] = bp[1];
    acc = __builtin_amdgcn_wmma_f32_16x16x32_bf16(false, a.v, false, b.v,
                                                  (short)0, acc, false, false);
  }
#pragma unroll 4
  for (int kb = kbx; kb < kbt; ++kb) {        // h part of A
    Frag a, b;
    const uint4* ap = (const uint4*)(hrow + (kb * 32 - in_dim) + hlf * 8);
    a.q[0] = ap[0]; a.q[1] = ap[2];
    const uint4* bp = (const uint4*)(Bzr + (((size_t)kb * 64 + nt) << 9) + (lane << 4));
    b.q[0] = bp[0]; b.q[1] = bp[1];
    acc = __builtin_amdgcn_wmma_f32_16x16x32_bf16(false, a.v, false, b.v,
                                                  (short)0, acc, false, false);
  }

  const int n = nt * 16 + l15;
  if (n < HID) {
    const float bias = bz[n];
#pragma unroll
    for (int v = 0; v < 8; ++v) {
      const int r_ = mt * 16 + v + hlf * 8;
      zbuf[r_ * HID + n] = sigmoidf_fast(acc[v] + bias);
    }
  } else {
    const int n2 = n - HID;
    const float bias = br[n2];
#pragma unroll
    for (int v = 0; v < 8; ++v) {
      const int r_ = mt * 16 + v + hlf * 8;
      const float rg = sigmoidf_fast(acc[v] + bias);
      rhbuf[r_ * HID + n2] = f2bf(rg * h_f32[r_ * HID + n2]);
    }
  }
}

// ---------------------------------------------------------------------------
// Phase 2 per timestep: [x_t , r*h] @ Bh -> hcan = tanh(. + bh);
// h_new = z*hcan + (1-z)*h_old.  Writes fp32 h, bf16 h, and the layer output
// (bf16 ping/pong for layers 0-1, fp32 d_out (+ final h) for layer 2).
// ---------------------------------------------------------------------------
__global__ __launch_bounds__(128)
void gru_phase2_kernel(const unsigned short* __restrict__ xin,
                       const unsigned short* __restrict__ rh_bf,
                       float* __restrict__ h_f32,
                       unsigned short* __restrict__ h_bf,
                       const float* __restrict__ zbuf,
                       const unsigned short* __restrict__ Bh,
                       const float* __restrict__ bh,
                       float* __restrict__ out_f32,
                       float* __restrict__ out_last,
                       unsigned short* __restrict__ out_bf,
                       int t, int in_dim, int is_last) {
  const int lane = threadIdx.x & 31;
  const int w = blockIdx.x * 4 + (threadIdx.x >> 5);
  const int mt = w & 3;
  const int nt = w >> 2;                      // 0..31
  const int hlf = lane >> 4;
  const int l15 = lane & 15;
  const int arow = mt * 16 + l15;
  const unsigned short* xrow = xin + ((size_t)arow * T_SZ + t) * in_dim;
  const unsigned short* hrow = rh_bf + (size_t)arow * HID;

  v8f acc = {};
  const int kbx = in_dim >> 5;
  const int kbt = (in_dim + HID) >> 5;
#pragma unroll 4
  for (int kb = 0; kb < kbx; ++kb) {
    Frag a, b;
    const uint4* ap = (const uint4*)(xrow + kb * 32 + hlf * 8);
    a.q[0] = ap[0]; a.q[1] = ap[2];
    const uint4* bp = (const uint4*)(Bh + (((size_t)kb * 32 + nt) << 9) + (lane << 4));
    b.q[0] = bp[0]; b.q[1] = bp[1];
    acc = __builtin_amdgcn_wmma_f32_16x16x32_bf16(false, a.v, false, b.v,
                                                  (short)0, acc, false, false);
  }
#pragma unroll 4
  for (int kb = kbx; kb < kbt; ++kb) {
    Frag a, b;
    const uint4* ap = (const uint4*)(hrow + (kb * 32 - in_dim) + hlf * 8);
    a.q[0] = ap[0]; a.q[1] = ap[2];
    const uint4* bp = (const uint4*)(Bh + (((size_t)kb * 32 + nt) << 9) + (lane << 4));
    b.q[0] = bp[0]; b.q[1] = bp[1];
    acc = __builtin_amdgcn_wmma_f32_16x16x32_bf16(false, a.v, false, b.v,
                                                  (short)0, acc, false, false);
  }

  const int n = nt * 16 + l15;
  const float bias = bh[n];
#pragma unroll
  for (int v = 0; v < 8; ++v) {
    const int r_ = mt * 16 + v + hlf * 8;
    const int o = r_ * HID + n;
    const float hcan = tanhf_fast(acc[v] + bias);
    const float hold = h_f32[o];
    const float z = zbuf[o];
    const float hnew = z * hcan + (1.0f - z) * hold;
    h_f32[o] = hnew;
    h_bf[o] = f2bf(hnew);
    const size_t oo = ((size_t)r_ * T_SZ + t) * HID + n;
    if (is_last) {
      out_f32[oo] = hnew;
      if (t == T_SZ - 1) out_last[o] = hnew;
    } else {
      out_bf[oo] = f2bf(hnew);
    }
  }
}

extern "C" void kernel_launch(void* const* d_in, const int* in_sizes, int n_in,
                              void* d_out, int out_size, void* d_ws, size_t ws_size,
                              hipStream_t stream) {
  (void)in_sizes; (void)n_in; (void)out_size; (void)ws_size;
  const float* x = (const float*)d_in[0];
  const float* prm[27];
  for (int i = 0; i < 27; ++i) prm[i] = (const float*)d_in[1 + i];

  char* p = (char*)d_ws;
  auto wsalloc = [&](size_t bytes) -> void* {
    void* r = (void*)p;
    p += (bytes + 255) & ~(size_t)255;
    return r;
  };
  unsigned short* xbuf0 = (unsigned short*)wsalloc((size_t)B_SZ * T_SZ * IN0 * 2);
  unsigned short* ping  = (unsigned short*)wsalloc((size_t)B_SZ * T_SZ * HID * 2);
  unsigned short* pong  = (unsigned short*)wsalloc((size_t)B_SZ * T_SZ * HID * 2);
  unsigned short* Bzr[3];
  unsigned short* Bhp[3];
  for (int L = 0; L < 3; ++L) {
    const int K = (L ? HID : IN0) + HID;
    Bzr[L] = (unsigned short*)wsalloc((size_t)K * 1024 * 2);
    Bhp[L] = (unsigned short*)wsalloc((size_t)K * 512 * 2);
  }
  float* h_f32          = (float*)wsalloc((size_t)B_SZ * HID * 4);
  unsigned short* h_bf  = (unsigned short*)wsalloc((size_t)B_SZ * HID * 2);
  float* zbuf           = (float*)wsalloc((size_t)B_SZ * HID * 4);
  unsigned short* rhbuf = (unsigned short*)wsalloc((size_t)B_SZ * HID * 2);

  // One-time per-call weight repack (bf16, WMMA-fragment order) + x conversion.
  for (int L = 0; L < 3; ++L) {
    const int in_dim = L ? HID : IN0;
    const long long tot = (long long)(in_dim + HID) * 1536;
    const int g = (int)((tot + 255) / 256);
    pack_weights_kernel<<<g, 256, 0, stream>>>(
        prm[9 * L + 0], prm[9 * L + 2],   // Wz, Uz
        prm[9 * L + 3], prm[9 * L + 5],   // Wr, Ur
        prm[9 * L + 6], prm[9 * L + 8],   // Wh, Uh
        in_dim, Bzr[L], Bhp[L]);
  }
  {
    const int n = B_SZ * T_SZ * IN0;
    convert_x_kernel<<<(n + 255) / 256, 256, 0, stream>>>(x, xbuf0, n);
  }

  float* out_seq  = (float*)d_out;
  float* out_last = out_seq + (size_t)B_SZ * T_SZ * HID;

  for (int L = 0; L < 3; ++L) {
    const int in_dim = L ? HID : IN0;
    const unsigned short* xin = (L == 0) ? xbuf0 : ((L == 1) ? ping : pong);
    unsigned short* obf = (L == 0) ? ping : pong;
    const int is_last = (L == 2);
    const float* bz = prm[9 * L + 1];
    const float* br = prm[9 * L + 4];
    const float* bh = prm[9 * L + 7];
    zero_h_kernel<<<(B_SZ * HID + 255) / 256, 256, 0, stream>>>(h_f32, h_bf);
    for (int t = 0; t < T_SZ; ++t) {
      gru_phase1_kernel<<<64, 128, 0, stream>>>(xin, h_bf, h_f32, Bzr[L], bz, br,
                                                zbuf, rhbuf, t, in_dim);
      gru_phase2_kernel<<<32, 128, 0, stream>>>(xin, rhbuf, h_f32, h_bf, zbuf,
                                                Bhp[L], bh, out_seq, out_last, obf,
                                                t, in_dim, is_last);
    }
  }
}
